// SA_MH_Layer_81123342287376
// MI455X (gfx1250) — matrix operand order repaired and checked by reading, not compiled, not run on hardware
//
#include <hip/hip_runtime.h>

typedef __bf16 bf16;
typedef __attribute__((ext_vector_type(16))) __bf16 v16bf;
typedef __attribute__((ext_vector_type(8)))  __bf16 v8bf;
typedef __attribute__((ext_vector_type(4)))  __bf16 v4bf;
typedef __attribute__((ext_vector_type(2)))  __bf16 v2bf;
typedef __attribute__((ext_vector_type(8)))  float  v8f;
typedef __attribute__((ext_vector_type(4)))  int    v4i;

// LDS (AS3)-qualified types: guarantee ds_* instructions for all LDS traffic
typedef __attribute__((address_space(3))) __bf16 ldsbf;
typedef __attribute__((address_space(3))) v8bf   av8bf;

#define BSZ 128
#define CH  512
#define PP  512
#define HH  4
#define HD  128        // head dim

#define MT 128
#define NT 128
#define KT 32
#define LDS_K 40       // KT + 8 pad  -> 80B row stride (16B aligned)
#define LDS_Q 136      // 128 + 8 pad -> 272B row stride (16B aligned)
#define LDS_S 520      // 512 + 8 pad -> 1040B row stride (16B aligned)
#define TILE_A (MT * LDS_K)

// ---------------- CDNA5 async global->LDS staging ----------------

#if defined(__AMDGCN__) && __has_builtin(__builtin_amdgcn_global_load_async_to_lds_b128)
#define HAS_ASYNC 1
typedef __attribute__((address_space(1))) v4i* g_v4i_p;   // global int4*
typedef __attribute__((address_space(3))) v4i* l_v4i_p;   // LDS int4*
#endif

__device__ __forceinline__ void copy16_g2l(const bf16* g, ldsbf* l) {
#if defined(HAS_ASYNC)
  __builtin_amdgcn_global_load_async_to_lds_b128((g_v4i_p)g, (l_v4i_p)l, 0, 0);
#else
  *(av8bf*)l = *(const v8bf*)g;
#endif
}

__device__ __forceinline__ void wait_staging() {
#if defined(HAS_ASYNC)
#if __has_builtin(__builtin_amdgcn_s_wait_asynccnt)
  __builtin_amdgcn_s_wait_asynccnt(0);
#else
  asm volatile("s_wait_asynccnt 0x0" ::: "memory");
#endif
#endif
}

// 128 rows x 32 k (K-contiguous) -> LDS stride LDS_K. 2 x b128 per thread.
__device__ __forceinline__ void stage_tile(ldsbf* dst, const bf16* src, int ld, int tid) {
#pragma unroll
  for (int i = 0; i < 2; ++i) {
    int e = (i * 256 + tid) * 8;          // 4096 elems
    int m = e >> 5;
    int k = e & 31;
    copy16_g2l(src + m * ld + k, dst + m * LDS_K + k);
  }
}

// 128 rows x 128 k -> LDS stride LDS_Q. 8 x b128 per thread.
__device__ __forceinline__ void stage_q(ldsbf* dst, const bf16* src, int ld, int tid) {
#pragma unroll
  for (int i = 0; i < 8; ++i) {
    int e = (i * 256 + tid) * 8;          // 16384 elems
    int m = e >> 7;
    int k = e & 127;
    copy16_g2l(src + m * ld + k, dst + m * LDS_Q + k);
  }
}

// ---------------- fragment helpers (ISA 7.12.2 layouts) ----------------

__device__ __forceinline__ v16bf frag_join(v8bf lo, v8bf hi) {
  v16bf r;
#pragma unroll
  for (int i = 0; i < 8; ++i) { r[i] = lo[i]; r[i + 8] = hi[i]; }
  return r;
}

// A 16x32: lane lo = row M; K = 8*hi + j (j<8), 16 + 8*hi + (j-8)
__device__ __forceinline__ v16bf load_frag_a(const ldsbf* sA, int lda, int m, int hi) {
  const ldsbf* p = sA + m * lda + 8 * hi;
  v8bf lo8 = *(const av8bf*)p;
  v8bf hi8 = *(const av8bf*)(p + 16);
  return frag_join(lo8, hi8);
}

// B 32x16 staged K-contiguous per n (Bt[n][k]): lane lo = N; K = 8*hi + j (j<8), 16 + 8*hi + (j-8)
__device__ __forceinline__ v16bf load_frag_b(const ldsbf* sBt, int ldb, int n, int hi) {
  const ldsbf* p = sBt + n * ldb + 8 * hi;
  v8bf lo8 = *(const av8bf*)p;
  v8bf hi8 = *(const av8bf*)(p + 16);
  return frag_join(lo8, hi8);
}

// ---------------- one 32-deep K step: 8 WMMAs per wave ----------------

__device__ __forceinline__ void mma_step(const ldsbf* sA, int lda,
                                         const ldsbf* sBt, int ldb,
                                         v8f (&acc)[2][4], int wm, int wn, int lane) {
  const int lo = lane & 15, hi = lane >> 4;
  v16bf a[2], b[4];
#pragma unroll
  for (int i = 0; i < 2; ++i) a[i] = load_frag_a(sA, lda, wm * 32 + i * 16 + lo, hi);
#pragma unroll
  for (int j = 0; j < 4; ++j) b[j] = load_frag_b(sBt, ldb, wn * 64 + j * 16 + lo, hi);
#pragma unroll
  for (int i = 0; i < 2; ++i)
#pragma unroll
    for (int j = 0; j < 4; ++j)
      acc[i][j] = __builtin_amdgcn_wmma_f32_16x16x32_bf16(
          false, a[i], false, b[j], (short)0, acc[i][j], false, false);
}

// D layout: VGPR r: m = mb + r + 8*hi, n = nb + (lane&15)
__device__ __forceinline__ void store_trans_bf16(bf16* out, int ld, const v8f& a,
                                                 int mb, int nb, int lane) {
  const int lo = lane & 15, hi = lane >> 4;
  v8bf p;
#pragma unroll
  for (int r = 0; r < 8; ++r) p[r] = (bf16)a[r];
  *(v8bf*)(out + (nb + lo) * ld + mb + 8 * hi) = p;      // 16B aligned store
}

__device__ __forceinline__ void store_direct_bf16(bf16* out, int ld, const v8f& a,
                                                  int mb, int nb, int lane) {
  const int lo = lane & 15, hi = lane >> 4;
#pragma unroll
  for (int r = 0; r < 8; ++r) out[(mb + 8 * hi + r) * ld + nb + lo] = (bf16)a[r];
}

// ---------------- kernels ----------------

// blocks [0, 32768): tiled transpose x (b,f,p) f32 -> xt (b,p,f) bf16
// blocks [32768, 33792): convert the four 512x512 weights to bf16
__global__ void __launch_bounds__(256) cvt_kernel(const float* __restrict__ x,
                                                  const float* __restrict__ Wq,
                                                  const float* __restrict__ Wk,
                                                  const float* __restrict__ Wv,
                                                  const float* __restrict__ Wt,
                                                  bf16* __restrict__ xt,
                                                  bf16* __restrict__ wbf) {
  __shared__ float tile[32][33];
  const int tid = threadIdx.x;
  if (blockIdx.x < 32768) {
    int bb = blockIdx.x;
    int b  = bb >> 8;
    int tf = (bb & 255) >> 4;             // f tile
    int tp = bb & 15;                     // p tile
    const float* xb = x + (b * CH + tf * 32) * PP + tp * 32;
#pragma unroll
    for (int i = 0; i < 4; ++i) {
      int r = (tid >> 5) + i * 8;
      int c = tid & 31;
      tile[r][c] = xb[r * PP + c];
    }
    __syncthreads();
    bf16* ob = xt + (b * PP + tp * 32) * CH + tf * 32;
#pragma unroll
    for (int i = 0; i < 2; ++i) {
      int q  = i * 256 + tid;             // 512 pairs
      int c  = q >> 4;                    // p within tile
      int r2 = (q & 15) * 2;              // f within tile (pair)
      v2bf pr;
      pr[0] = (bf16)tile[r2][c];
      pr[1] = (bf16)tile[r2 + 1][c];
      *(v2bf*)(ob + c * CH + r2) = pr;
    }
  } else {
    int i = ((blockIdx.x - 32768) * 256 + tid) * 4;      // < 4*262144
    const float* src; int k = i;
    if      (i <  262144) { src = Wq; }
    else if (i <  524288) { src = Wk; k -= 262144; }
    else if (i <  786432) { src = Wv; k -= 524288; }
    else                  { src = Wt; k -= 786432; }
    float4 f = *(const float4*)(src + k);
    v4bf o; o[0] = (bf16)f.x; o[1] = (bf16)f.y; o[2] = (bf16)f.z; o[3] = (bf16)f.w;
    *(v4bf*)(wbf + i) = o;
  }
}

// O[b] = W (512x512) x X[b];  A = W[o][f], B = xt[b][p][f] ([n][k] row-major).
// TRANS=1 stores O^T (Qt[p][o], Kt[q][c]); TRANS=0 stores direct (Vm[c][q]).
template <int TRANS>
__global__ void __launch_bounds__(256) gemm_qkv(const bf16* __restrict__ W,
                                                const bf16* __restrict__ Xt,
                                                bf16* __restrict__ O) {
  __shared__ bf16 sAm[2 * TILE_A];
  __shared__ bf16 sBm[2 * TILE_A];
  ldsbf* sA0 = (ldsbf*)sAm;
  ldsbf* sB0 = (ldsbf*)sBm;
  const int b = blockIdx.z, mt = blockIdx.y, nt = blockIdx.x;
  const int tid = threadIdx.x, lane = tid & 31, wave = tid >> 5;
  const int wm = wave & 3, wn = wave >> 2;
  const bf16* Ab = W  + (mt * MT) * CH;
  const bf16* Bb = Xt + b * (CH * PP) + (nt * NT) * CH;
  bf16* Ob = O + b * (CH * PP);
  v8f acc[2][4] = {};
  stage_tile(sA0, Ab, CH, tid);
  stage_tile(sB0, Bb, CH, tid);
#pragma unroll 1
  for (int kb = 0; kb < CH / KT; ++kb) {
    int cur = (kb & 1) * TILE_A, nxt = TILE_A - cur;
    wait_staging();
    __syncthreads();
    if (kb + 1 < CH / KT) {
      stage_tile(sA0 + nxt, Ab + (kb + 1) * KT, CH, tid);
      stage_tile(sB0 + nxt, Bb + (kb + 1) * KT, CH, tid);
    }
    mma_step(sA0 + cur, LDS_K, sB0 + cur, LDS_K, acc, wm, wn, lane);
    __syncthreads();
  }
#pragma unroll
  for (int i = 0; i < 2; ++i)
#pragma unroll
    for (int j = 0; j < 4; ++j) {
      int mb = mt * MT + wm * 32 + i * 16;
      int nb = nt * NT + wn * 64 + j * 16;
      if (TRANS) store_trans_bf16 (Ob, PP, acc[i][j], mb, nb, lane);
      else       store_direct_bf16(Ob, PP, acc[i][j], mb, nb, lane);
    }
}

// Fused attention per (b, h, 128-row l-block):
//   S[l][q] = Qt_slice x Kt_slice  (128x512 bf16, LDS-resident)
//   U[l][d] = S x Vm_slice         -> direct store Tt[b][l][c]
__global__ void __launch_bounds__(256) attn_fused(const bf16* __restrict__ Qt,
                                                  const bf16* __restrict__ Kt,
                                                  const bf16* __restrict__ Vm,
                                                  bf16* __restrict__ Tt) {
  extern __shared__ bf16 smem[];
  ldsbf* sS = (ldsbf*)smem;                 // 128 * LDS_S
  ldsbf* sQ = sS + MT * LDS_S;              // 128 * LDS_Q
  ldsbf* sB0 = sQ + MT * LDS_Q;             // 2 x 128 * LDS_K
  const int lb = blockIdx.x, h = blockIdx.y, b = blockIdx.z;
  const int tid = threadIdx.x, lane = tid & 31, wave = tid >> 5;
  const int wm = wave & 3, wn = wave >> 2;
  const int lo = lane & 15, hi = lane >> 4;
  const bf16* Qb = Qt + b * (CH * PP);      // [l][c]
  const bf16* Kb = Kt + b * (CH * PP);      // [q][c]
  const bf16* Vb = Vm + b * (CH * PP) + (h * HD) * PP;   // [d][q]

  // preload 128x128 Q slice once
  stage_q(sQ, Qb + (lb * MT) * CH + h * HD, CH, tid);

  // ---- GEMM1: S[l=128][q=512], 16 pipelined 32-deep steps ----
  stage_tile(sB0, Kb + h * HD, CH, tid);
  v8f acc[2][4] = {};
#pragma unroll 1
  for (int t = 0; t < 16; ++t) {
    int nc = t >> 2, kb = t & 3;
    int cur = (t & 1) * TILE_A, nxt = TILE_A - cur;
    wait_staging();
    __syncthreads();
    if (t + 1 < 16) {
      int nc2 = (t + 1) >> 2, kb2 = (t + 1) & 3;
      stage_tile(sB0 + nxt, Kb + (nc2 * NT) * CH + h * HD + kb2 * KT, CH, tid);
    }
    mma_step(sQ + kb * KT, LDS_Q, sB0 + cur, LDS_K, acc, wm, wn, lane);
    __syncthreads();
    if (kb == 3) {                          // flush this 128x128 S chunk to LDS
#pragma unroll
      for (int i = 0; i < 2; ++i)
#pragma unroll
        for (int j = 0; j < 4; ++j) {
          int mb = wm * 32 + i * 16;
          int nb = nc * NT + wn * 64 + j * 16;
#pragma unroll
          for (int r = 0; r < 8; ++r)
            sS[(mb + 8 * hi + r) * LDS_S + nb + lo] = (bf16)acc[i][j][r];
          acc[i][j] = (v8f)(0.0f);
        }
    }
  }

  // ---- GEMM2: U[l=128][d=128], K = 512 over q, A straight from sS ----
  stage_tile(sB0, Vb, PP, tid);
  v8f acc2[2][4] = {};
#pragma unroll 1
  for (int kb = 0; kb < PP / KT; ++kb) {
    int cur = (kb & 1) * TILE_A, nxt = TILE_A - cur;
    wait_staging();
    __syncthreads();
    if (kb + 1 < PP / KT) stage_tile(sB0 + nxt, Vb + (kb + 1) * KT, PP, tid);
    mma_step(sS + kb * KT, LDS_S, sB0 + cur, LDS_K, acc2, wm, wn, lane);
    __syncthreads();
  }
  bf16* Tb = Tt + b * (CH * PP);            // [l][c]
#pragma unroll
  for (int i = 0; i < 2; ++i)
#pragma unroll
    for (int j = 0; j < 4; ++j) {
      int mb = lb * MT + wm * 32 + i * 16;  // l
      int nb = h * HD + wn * 64 + j * 16;   // c
      store_direct_bf16(Tb, CH, acc2[i][j], mb, nb, lane);
    }
}

// y[b] = Wt x T[b] + bt - x[b]   (fp32, exact BN stats downstream)
// A = Wt[o][c], B = Tt[b][l][c] ([n][k] row-major)
__global__ void __launch_bounds__(256) out_gemm(const bf16* __restrict__ Wt,
                                                const bf16* __restrict__ Tt,
                                                const float* __restrict__ bt,
                                                const float* __restrict__ x,
                                                float* __restrict__ y) {
  __shared__ bf16 sAm[2 * TILE_A];
  __shared__ bf16 sBm[2 * TILE_A];
  ldsbf* sA0 = (ldsbf*)sAm;
  ldsbf* sB0 = (ldsbf*)sBm;
  const int b = blockIdx.z, mt = blockIdx.y, nt = blockIdx.x;
  const int tid = threadIdx.x, lane = tid & 31, wave = tid >> 5;
  const int wm = wave & 3, wn = wave >> 2;
  const int lo = lane & 15, hi = lane >> 4;
  const bf16* Ab = Wt + (mt * MT) * CH;
  const bf16* Bb = Tt + b * (CH * PP) + (nt * NT) * CH;
  v8f acc[2][4] = {};
  stage_tile(sA0, Ab, CH, tid);
  stage_tile(sB0, Bb, CH, tid);
#pragma unroll 1
  for (int kb = 0; kb < CH / KT; ++kb) {
    int cur = (kb & 1) * TILE_A, nxt = TILE_A - cur;
    wait_staging();
    __syncthreads();
    if (kb + 1 < CH / KT) {
      stage_tile(sA0 + nxt, Ab + (kb + 1) * KT, CH, tid);
      stage_tile(sB0 + nxt, Bb + (kb + 1) * KT, CH, tid);
    }
    mma_step(sA0 + cur, LDS_K, sB0 + cur, LDS_K, acc, wm, wn, lane);
    __syncthreads();
  }
  const float* xb = x + b * (CH * PP);
  float* yb = y + b * (CH * PP);
#pragma unroll
  for (int i = 0; i < 2; ++i)
#pragma unroll
    for (int j = 0; j < 4; ++j) {
      int mb = mt * MT + wm * 32 + i * 16;
      int nb = nt * NT + wn * 64 + j * 16;
      int n  = nb + lo;
#pragma unroll
      for (int r = 0; r < 8; ++r) {
        int m = mb + 8 * hi + r;
        yb[m * PP + n] = acc[i][j][r] + bt[m] - xb[m * PP + n];
      }
    }
}

// per-channel sum / sumsq over (batch, length); one block per channel, no atomics
__global__ void __launch_bounds__(256) bn_reduce(const float* __restrict__ y,
                                                 float* __restrict__ sums) {
  __shared__ float s1[256], s2[256];
  const int o = blockIdx.x, tid = threadIdx.x;
  float a = 0.f, b = 0.f;
#pragma unroll 1
  for (int bb = 0; bb < BSZ; ++bb) {
    const float* row = y + (bb * CH + o) * PP;
    for (int l = tid; l < PP; l += 256) { float v = row[l]; a += v; b += v * v; }
  }
  s1[tid] = a; s2[tid] = b;
  __syncthreads();
  for (int s = 128; s > 0; s >>= 1) {
    if (tid < s) { s1[tid] += s1[tid + s]; s2[tid] += s2[tid + s]; }
    __syncthreads();
  }
  if (tid == 0) { sums[o] = s1[0]; sums[CH + o] = s2[0]; }
}

// out = x + relu(gamma*(y-mean)*rsqrt(var+eps)+beta)
__global__ void __launch_bounds__(256) bn_finish(const float* __restrict__ y,
                                                 const float* __restrict__ x,
                                                 const float* __restrict__ sums,
                                                 const float* __restrict__ gamma,
                                                 const float* __restrict__ beta,
                                                 float* __restrict__ out) {
  int i = (blockIdx.x * 256 + threadIdx.x) * 4;
  int o = (i >> 9) & (CH - 1);
  const float rn   = 1.0f / (float)(BSZ * PP);
  float mean  = sums[o] * rn;
  float var   = sums[CH + o] * rn - mean * mean;
  float scale = gamma[o] * rsqrtf(var + 1e-5f);
  float bias  = beta[o];
  float4 yv = *(const float4*)(y + i);
  float4 xv = *(const float4*)(x + i);
  float4 ov;
  ov.x = xv.x + fmaxf(scale * (yv.x - mean) + bias, 0.f);
  ov.y = xv.y + fmaxf(scale * (yv.y - mean) + bias, 0.f);
  ov.z = xv.z + fmaxf(scale * (yv.z - mean) + bias, 0.f);
  ov.w = xv.w + fmaxf(scale * (yv.w - mean) + bias, 0.f);
  *(float4*)(out + i) = ov;
}

// ---------------- host launcher ----------------

extern "C" void kernel_launch(void* const* d_in, const int* in_sizes, int n_in,
                              void* d_out, int out_size, void* d_ws, size_t ws_size,
                              hipStream_t stream) {
  const float* x     = (const float*)d_in[0];
  const float* Wq    = (const float*)d_in[1];
  const float* Wk    = (const float*)d_in[2];
  const float* Wv    = (const float*)d_in[3];
  const float* Wt    = (const float*)d_in[4];
  const float* bt    = (const float*)d_in[5];
  const float* gamma = (const float*)d_in[6];
  const float* beta  = (const float*)d_in[7];
  float* out = (float*)d_out;

  const size_t NXE = (size_t)BSZ * CH * PP;   // 33,554,432 elems
  char* p = (char*)d_ws;
  bf16*  xt   = (bf16*)p;  p += NXE * 2;                 // x^T  [b][p][f]
  bf16*  wbf  = (bf16*)p;  p += (size_t)4 * CH * CH * 2;
  bf16*  Qt   = (bf16*)p;  p += NXE * 2;                 // Q^T  [b][l][c]
  bf16*  Kt   = (bf16*)p;  p += NXE * 2;                 // K^T  [b][q][c]
  bf16*  Vm   = (bf16*)p;  p += NXE * 2;                 // V    [b][c][q]
  bf16*  Tt   = (bf16*)p;  p += NXE * 2;                 // T^T  [b][l][c]
  float* yv   = (float*)p; p += NXE * 4;
  float* sums = (float*)p; p += 2 * CH * 4;
  (void)ws_size; (void)in_sizes; (void)n_in; (void)out_size;

  // 1) transpose+convert x, convert weights
  cvt_kernel<<<32768 + 1024, 256, 0, stream>>>(x, Wq, Wk, Wv, Wt, xt, wbf);

  dim3 g(PP / NT, CH / MT, BSZ);                         // (4,4,128)

  // 2) QKV projections: Q,K transposed; V direct (all downstream reads row-major)
  gemm_qkv<1><<<g, 256, 0, stream>>>(wbf + 0 * CH * CH, xt, Qt);
  gemm_qkv<1><<<g, 256, 0, stream>>>(wbf + 1 * CH * CH, xt, Kt);
  gemm_qkv<0><<<g, 256, 0, stream>>>(wbf + 2 * CH * CH, xt, Vm);

  // 3) fused attention (S LDS-resident; ~184 KiB dynamic LDS)
  {
    size_t smem = (size_t)(MT * LDS_S + MT * LDS_Q + 2 * TILE_A) * sizeof(bf16);
    (void)hipFuncSetAttribute((const void*)attn_fused,
                              hipFuncAttributeMaxDynamicSharedMemorySize, (int)smem);
    attn_fused<<<g, 256, smem, stream>>>(Qt, Kt, Vm, Tt);
  }

  // 4) trans_conv + bias - residual -> y (fp32)
  out_gemm<<<g, 256, 0, stream>>>(wbf + 3 * CH * CH, Tt, bt, x, yv);

  // 5) deterministic batch-norm statistics
  bn_reduce<<<CH, 256, 0, stream>>>(yv, sums);

  // 6) normalize + relu + residual
  bn_finish<<<(int)(NXE / 1024), 256, 0, stream>>>(yv, x, sums, gamma, beta, out);
}
